// Model4_24739011625470
// MI455X (gfx1250) — matrix operand order repaired and checked by reading, not compiled, not run on hardware
//
#include <hip/hip_runtime.h>
#include <stddef.h>

typedef __attribute__((ext_vector_type(16))) _Float16 v16h;
typedef __attribute__((ext_vector_type(8)))  _Float16 v8h;
typedef __attribute__((ext_vector_type(4)))  _Float16 v4h;
typedef __attribute__((ext_vector_type(8)))  float    v8f;

#define NEG_SLOPE 0.01f

// K ordering for the implicit GEMM (both A and B sides):
//   k = (kh*KWP + kw)*CINP + c,  c in [0,CINP), kw in [0,KWP)
// CINP/KWP are zero-padded so every aligned run of 8 consecutive k maps to
// contiguous LDS addresses in an HWC patch; padded positions carry ZERO
// weights in B, so the A side needs no guards anywhere.

// ---------------------------------------------------------------------------
// Pack conv weights (OIHW fp32) into WMMA B-fragment order, fp16.
// B is 32x16 (KxN), packed in this code's contiguous k order: lane l<16 -> N=l, K=slot s; lane l>=16 -> N=l-16, K=16+s.
// Layout: bpack[((g*KC + kc)*32 + lane)*16 + s]; padded k slots are zero.
// ---------------------------------------------------------------------------
template<int COUT, int CIN, int KH, int KW, int CINP, int KWP>
__global__ void pack_weights_kernel(const float* __restrict__ W,
                                    _Float16* __restrict__ bpack) {
  constexpr int KP = KH * KWP * CINP;
  constexpr int KC = (KP + 31) / 32;
  constexpr int G  = COUT / 16;
  const int total = G * KC * 32 * 16;
  int idx = blockIdx.x * blockDim.x + threadIdx.x;
  if (idx >= total) return;
  const int s  = idx & 15;
  const int l  = (idx >> 4) & 31;
  const int u  = idx >> 9;
  const int kc = u % KC;
  const int g  = u / KC;
  const int k  = kc * 32 + ((l < 16) ? s : 16 + s);
  const int n  = g * 16 + (l & 15);
  const int p  = k / CINP;
  const int c  = k - p * CINP;
  const int kh = p / KWP;
  const int kw = p - kh * KWP;
  float v = 0.0f;
  if (kh < KH && kw < KW && c < CIN)
    v = W[((n * CIN + c) * KH + kh) * KW + kw];
  bpack[idx] = (_Float16)v;
}

// ---------------------------------------------------------------------------
// Fused conv (implicit GEMM via v_wmma_f32_16x16x32_f16) + bias + maxpool +
// LeakyReLU.  One block = (image n, pooled tile TPHxTPW, 16-channel group g).
//
// Hot loop per wave (kc outer, M-tiles inner, accs resident in VGPRs):
//   - ktab[k] = (kh*PPW + kw)*CINP + c run-start offsets (u16, LDS);
//     2 scalar table loads per kc; within-run offsets are +1 by construction.
//   - A fragment = 2 x ds_load_b128 (CINP%8==0) or 4 x ds_load_b64 (CINP=4).
//   - B fragment loads once per kc from global, reused by all M tiles.
//   - no exec-mask guards: every wave runs MMAX tiles on clamped pixels;
//     only ctile stores are bounds-checked.
// ---------------------------------------------------------------------------
template<int CIN, int CINP, int KWP, int G, int KH, int KW, int PAD,
         int HIN, int WIN, int POOLK, int POOLS, int HPOOL, int WPOOL,
         int TPH, int TPW, typename InT, typename OutT>
__global__ __launch_bounds__(256) void conv_pool_wmma_kernel(
    const InT* __restrict__ in, const _Float16* __restrict__ bpack,
    const float* __restrict__ bias, OutT* __restrict__ out) {
  constexpr int COUT = G * 16;
  constexpr int KP   = KH * KWP * CINP;
  constexpr int KC   = (KP + 31) / 32;
  constexpr int CH   = (TPH - 1) * POOLS + POOLK;   // conv rows needed
  constexpr int CW   = (TPW - 1) * POOLS + POOLK;   // conv cols needed
  constexpr int PPH  = CH + KH - 1;                 // patch rows
  constexpr int PPW  = CW + KWP - 1;                // patch cols (kw-padded)
  constexpr int CPX  = CH * CW;                     // conv pixels (GEMM M)
  constexpr int MT   = (CPX + 15) / 16;             // M tiles of 16
  constexpr int NW   = 256 / 32;                    // waves per block (wave32)
  constexpr int MMAX = (MT + NW - 1) / NW;          // M tiles per wave

  __shared__ _Float16 patch[PPH * PPW * CINP];      // HWC, channel-padded
  __shared__ float    ctile[CPX * 16];
  __shared__ unsigned short ktab[KC * 32];

  const int n   = blockIdx.x;
  const int ty  = blockIdx.y;
  const int tx  = blockIdx.z / G;
  const int g   = blockIdx.z % G;
  const int tid = threadIdx.x;
  const int y0  = ty * TPH * POOLS;                 // conv-row origin of tile
  const int x0  = tx * TPW * POOLS;                 // conv-col origin of tile

  // ---- k -> patch-offset table (divisions once per block) ----
  for (int k = tid; k < KC * 32; k += 256) {
    const int p  = k / CINP;
    const int c  = k - p * CINP;
    const int kh = p / KWP;
    const int kw = p - kh * KWP;
    ktab[k] = (kh < KH) ? (unsigned short)((kh * PPW + kw) * CINP + c) : 0;
  }

  // ---- stage input patch, HWC (global reads stay row-contiguous) ----
  for (int e = tid; e < CIN * PPH * PPW; e += 256) {
    const int c  = e / (PPH * PPW);
    const int r  = e - c * (PPH * PPW);
    const int iy = r / PPW;
    const int ix = r - iy * PPW;
    const int gy = y0 + iy - PAD;
    const int gx = x0 + ix - PAD;
    float v = 0.0f;
    if (gy >= 0 && gy < HIN && gx >= 0 && gx < WIN)
      v = (float)in[((n * CIN + c) * HIN + gy) * WIN + gx];
    patch[(iy * PPW + ix) * CINP + c] = (_Float16)v;
  }
  if constexpr (CINP > CIN) {   // zero the padded channels
    for (int e = tid; e < (CINP - CIN) * PPH * PPW; e += 256) {
      const int c  = CIN + e / (PPH * PPW);
      const int r  = e - (e / (PPH * PPW)) * (PPH * PPW);
      patch[r * CINP + c] = (_Float16)0.0f;
    }
  }
  __syncthreads();

  const int lane = tid & 31;
  const int wid  = tid >> 5;
  const int krun = (lane < 16) ? 0 : 16;  // A 16x32 layout lane split

  int pixc[MMAX];          // pixel base offset * CINP (element units)
  v8f acc[MMAX];
#pragma unroll
  for (int i = 0; i < MMAX; ++i) {
    int m = (wid + i * NW) * 16 + (lane & 15);
    if (m >= CPX) m = CPX - 1;           // clamp; stores are guarded below
    const int oy = m / CW;
    pixc[i] = (oy * PPW + (m - oy * CW)) * CINP;
    v8f z = {};
    acc[i] = z;
  }

  for (int kc = 0; kc < KC; ++kc) {
    const int kb = kc * 32 + krun;
    const int tA = (int)ktab[kb];        // run starts; +1.. +7 implicit
    const int tB = (int)ktab[kb + 8];
    // B fragment: 32 contiguous bytes per lane, shared by all M tiles
    const v16h b =
        *(const v16h*)(bpack + ((((size_t)g * KC + kc) * 32 + lane) << 4));
#pragma unroll
    for (int i = 0; i < MMAX; ++i) {
      v16h a;
      if constexpr (CINP % 8 == 0) {
        const v8h lo = *(const v8h*)(patch + pixc[i] + tA);
        const v8h hi = *(const v8h*)(patch + pixc[i] + tB);
#pragma unroll
        for (int s = 0; s < 8; ++s) { a[s] = lo[s]; a[8 + s] = hi[s]; }
      } else {
        const v4h l0 = *(const v4h*)(patch + pixc[i] + tA);
        const v4h l1 = *(const v4h*)(patch + pixc[i] + tA + 4);
        const v4h h0 = *(const v4h*)(patch + pixc[i] + tB);
        const v4h h1 = *(const v4h*)(patch + pixc[i] + tB + 4);
#pragma unroll
        for (int s = 0; s < 4; ++s) {
          a[s] = l0[s]; a[4 + s] = l1[s]; a[8 + s] = h0[s]; a[12 + s] = h1[s];
        }
      }
      acc[i] = __builtin_amdgcn_wmma_f32_16x16x32_f16(
          /*neg_a=*/false, a, /*neg_b=*/false, b,
          /*c_mod=*/(short)0, acc[i], /*reuse_a=*/false, /*reuse_b=*/false);
    }
  }

  // C/D layout: VGPR v -> M = mt*16 + v (+8 for lanes 16..31), N = lane&15
#pragma unroll
  for (int i = 0; i < MMAX; ++i) {
#pragma unroll
    for (int v = 0; v < 8; ++v) {
      const int mm = (wid + i * NW) * 16 + v + ((lane >> 4) << 3);
      if (mm < CPX) ctile[mm * 16 + (lane & 15)] = acc[i][v];
    }
  }
  __syncthreads();

  // ---- maxpool + bias + LeakyReLU (bias commutes with max) ----
  for (int idx = tid; idx < TPH * TPW * 16; idx += 256) {
    const int ch = idx & 15;
    const int p  = idx >> 4;
    const int py = p / TPW;
    const int px = p - py * TPW;
    const int gpy = ty * TPH + py;
    const int gpx = tx * TPW + px;
    if (gpy < HPOOL && gpx < WPOOL) {
      float mx = -3.0e38f;
#pragma unroll
      for (int ky = 0; ky < POOLK; ++ky)
#pragma unroll
        for (int kx = 0; kx < POOLK; ++kx) {
          const int m = (py * POOLS + ky) * CW + (px * POOLS + kx);
          mx = fmaxf(mx, ctile[m * 16 + ch]);
        }
      mx += bias[g * 16 + ch];
      const float r = (mx > 0.0f) ? mx : NEG_SLOPE * mx;
      out[(((size_t)n * COUT + (g * 16 + ch)) * HPOOL + gpy) * WPOOL + gpx] =
          (OutT)r;
    }
  }
}

// ---------------------------------------------------------------------------
// Attention tail: [8,30,256] -> 16-head attn (d_qk=2, d_v=4, softmax over
// the QUERY axis, per reference) -> [8,1920] -> Linear(10) -> softmax.
// ---------------------------------------------------------------------------
__global__ __launch_bounds__(256) void attn_tail_kernel(
    const float* __restrict__ Y, const float* __restrict__ Wq,
    const float* __restrict__ Wk, const float* __restrict__ Wv,
    const float* __restrict__ Wr, const float* __restrict__ br,
    float* __restrict__ out) {
  constexpr int T = 30, E = 256, H = 16;
  const int b = blockIdx.x, tid = threadIdx.x;
  __shared__ float sY[T * E];
  __shared__ float sQ[H * T * 2];
  __shared__ float sK[H * T * 2];
  __shared__ float sV[H * T * 4];
  __shared__ float sMx[H * T];
  __shared__ float sDen[H * T];
  __shared__ float sZ[T * 64];
  __shared__ float sO[10];

  for (int i = tid; i < T * E; i += 256) sY[i] = Y[b * T * E + i];
  __syncthreads();

  for (int i = tid; i < H * T; i += 256) {
    const int h = i / T, s = i - (i / T) * T;
    float q0 = 0, q1 = 0, k0 = 0, k1 = 0, v0 = 0, v1 = 0, v2 = 0, v3 = 0;
    for (int e = 0; e < E; ++e) {
      const float y = sY[s * E + e];
      q0 += y * Wq[(h * E + e) * 2 + 0];
      q1 += y * Wq[(h * E + e) * 2 + 1];
      k0 += y * Wk[(h * E + e) * 2 + 0];
      k1 += y * Wk[(h * E + e) * 2 + 1];
      v0 += y * Wv[(h * E + e) * 4 + 0];
      v1 += y * Wv[(h * E + e) * 4 + 1];
      v2 += y * Wv[(h * E + e) * 4 + 2];
      v3 += y * Wv[(h * E + e) * 4 + 3];
    }
    sQ[i * 2 + 0] = q0; sQ[i * 2 + 1] = q1;
    sK[i * 2 + 0] = k0; sK[i * 2 + 1] = k1;
    sV[i * 4 + 0] = v0; sV[i * 4 + 1] = v1;
    sV[i * 4 + 2] = v2; sV[i * 4 + 3] = v3;
  }
  __syncthreads();

  const float scale = 0.70710678118654752f;  // 1/sqrt(D_K), D_K=2
  for (int i = tid; i < H * T; i += 256) {   // column stats over query axis
    const int h = i / T, t = i - (i / T) * T;
    const float kt0 = sK[(h * T + t) * 2], kt1 = sK[(h * T + t) * 2 + 1];
    float mx = -3.0e38f;
    for (int s = 0; s < T; ++s) {
      const float w =
          (sQ[(h * T + s) * 2] * kt0 + sQ[(h * T + s) * 2 + 1] * kt1) * scale;
      mx = fmaxf(mx, w);
    }
    float den = 0.0f;
    for (int s = 0; s < T; ++s) {
      const float w =
          (sQ[(h * T + s) * 2] * kt0 + sQ[(h * T + s) * 2 + 1] * kt1) * scale;
      den += expf(w - mx);
    }
    sMx[i] = mx;
    sDen[i] = den;
  }
  __syncthreads();

  for (int i = tid; i < H * T; i += 256) {   // Z = alpha @ V
    const int h = i / T, s = i - (i / T) * T;
    const float qs0 = sQ[(h * T + s) * 2], qs1 = sQ[(h * T + s) * 2 + 1];
    float z0 = 0, z1 = 0, z2 = 0, z3 = 0;
    for (int t = 0; t < T; ++t) {
      const float w =
          (qs0 * sK[(h * T + t) * 2] + qs1 * sK[(h * T + t) * 2 + 1]) * scale;
      const float a = expf(w - sMx[h * T + t]) / sDen[h * T + t];
      z0 += a * sV[(h * T + t) * 4 + 0];
      z1 += a * sV[(h * T + t) * 4 + 1];
      z2 += a * sV[(h * T + t) * 4 + 2];
      z3 += a * sV[(h * T + t) * 4 + 3];
    }
    sZ[s * 64 + h * 4 + 0] = z0;
    sZ[s * 64 + h * 4 + 1] = z1;
    sZ[s * 64 + h * 4 + 2] = z2;
    sZ[s * 64 + h * 4 + 3] = z3;
  }
  __syncthreads();

  if (tid < 10) {
    float acc = br[tid];
    for (int j = 0; j < 1920; ++j) acc += Wr[tid * 1920 + j] * sZ[j];
    sO[tid] = acc;
  }
  __syncthreads();
  if (tid == 0) {
    float mx = sO[0];
    for (int o = 1; o < 10; ++o) mx = fmaxf(mx, sO[o]);
    float den = 0.0f, e[10];
    for (int o = 0; o < 10; ++o) { e[o] = expf(sO[o] - mx); den += e[o]; }
    for (int o = 0; o < 10; ++o) out[b * 10 + o] = e[o] / den;
  }
}

// ---------------------------------------------------------------------------
extern "C" void kernel_launch(void* const* d_in, const int* in_sizes, int n_in,
                              void* d_out, int out_size, void* d_ws,
                              size_t ws_size, hipStream_t stream) {
  const float* X  = (const float*)d_in[0];
  const float* W1 = (const float*)d_in[1];
  const float* b1 = (const float*)d_in[2];
  const float* W2 = (const float*)d_in[3];
  const float* b2 = (const float*)d_in[4];
  const float* W3 = (const float*)d_in[5];
  const float* b3 = (const float*)d_in[6];
  const float* Wq = (const float*)d_in[7];
  const float* Wk = (const float*)d_in[8];
  const float* Wv = (const float*)d_in[9];
  const float* Wr = (const float*)d_in[10];
  const float* br = (const float*)d_in[11];
  float* out = (float*)d_out;

  char* ws = (char*)d_ws;
  size_t off = 0;
  auto carve = [&](size_t bytes) -> char* {
    char* p = ws + off;
    off += (bytes + 255) & ~(size_t)255;
    return p;
  };
  // packed weights (fp16, WMMA B-fragment order); KC1=12, KC2=25, KC3=25
  _Float16* bp1 = (_Float16*)carve((size_t)1 * 12 * 512 * 2);  // 12 KB
  _Float16* bp2 = (_Float16*)carve((size_t)2 * 25 * 512 * 2);  // 50 KB
  _Float16* bp3 = (_Float16*)carve((size_t)4 * 25 * 512 * 2);  //100 KB
  // pooled activations
  _Float16* o1 = (_Float16*)carve((size_t)240 * 16 * 38 * 38 * 2);  // 11.1 MB
  _Float16* o2 = (_Float16*)carve((size_t)240 * 32 * 10 * 10 * 2);  //  1.5 MB
  float*    Yb = (float*)carve((size_t)240 * 256 * 4);              //  246 KB

  pack_weights_kernel<16, 3, 9, 9, 4, 10>
      <<<(1 * 12 * 512 + 255) / 256, 256, 0, stream>>>(W1, bp1);
  pack_weights_kernel<32, 16, 7, 7, 16, 7>
      <<<(2 * 25 * 512 + 255) / 256, 256, 0, stream>>>(W2, bp2);
  pack_weights_kernel<64, 32, 5, 5, 32, 5>
      <<<(4 * 25 * 512 + 255) / 256, 256, 0, stream>>>(W3, bp3);

  // conv1: 200x200 +pad1 9x9 -> 194x194, pool7s5 -> 38x38, 16ch (1 group)
  conv_pool_wmma_kernel<3, 4, 10, 1, 9, 9, 1, 200, 200, 7, 5, 38, 38, 2, 10,
                        float, _Float16>
      <<<dim3(240, 19, 4), 256, 0, stream>>>(X, bp1, b1, o1);
  // conv2: 38x38 7x7 -> 32x32, pool5s3 -> 10x10, 32ch (2 groups)
  conv_pool_wmma_kernel<16, 16, 7, 2, 7, 7, 0, 38, 38, 5, 3, 10, 10, 2, 10,
                        _Float16, _Float16>
      <<<dim3(240, 5, 2), 256, 0, stream>>>(o1, bp2, b2, o2);
  // conv3: 10x10 5x5 -> 6x6, pool3s2 -> 2x2, 64ch (4 groups); output fp32,
  // laid out so that [n, c*4 + py*2 + px] == reshape(B,T,256)
  conv_pool_wmma_kernel<32, 32, 5, 4, 5, 5, 0, 10, 10, 3, 2, 2, 2, 2, 2,
                        _Float16, float>
      <<<dim3(240, 1, 4), 256, 0, stream>>>(o2, bp3, b3, Yb);

  attn_tail_kernel<<<8, 256, 0, stream>>>(Yb, Wq, Wk, Wv, Wr, br, out);

  (void)in_sizes; (void)n_in; (void)out_size; (void)ws_size;
}
